// FeatureBlock_74328704024959
// MI455X (gfx1250) — matrix operand order repaired and checked by reading, not compiled, not run on hardware
//
#include <hip/hip_runtime.h>
#include <hip/hip_bf16.h>

typedef __attribute__((ext_vector_type(16))) _Float16 v16h;
typedef __attribute__((ext_vector_type(8)))  float    v8f;
typedef __attribute__((ext_vector_type(4)))  float    v4f;

#define NN 100000
#define NE 1600000
#define DD 128
#define HIDN 256
#define RADIN 64
#define SPHIN 32
#define NW 4   // waves per block

union V16H { v16h v; uint4 u[2]; };

// ---- B operand: packed f16, 512 elems per 32x16 chunk, lane-contiguous 32B ----
static __device__ __forceinline__ v16h ld_b(const _Float16* base, int chunk, int lane) {
  const uint4* p = reinterpret_cast<const uint4*>(base + (size_t)chunk * 512 + lane * 16);
  V16H t; t.u[0] = p[0]; t.u[1] = p[1];
  return t.v;
}

// ---- A operand from a row-major f32 row, NON-TEMPORAL (single-use streams) ----
static __device__ __forceinline__ v16h ld_a_f32_nt(const float* __restrict__ row, int koff, int hi) {
  const v4f* p0 = reinterpret_cast<const v4f*>(row + koff + hi * 8);
  const v4f* p1 = reinterpret_cast<const v4f*>(row + koff + 16 + hi * 8);
  v4f a = __builtin_nontemporal_load(p0);
  v4f b = __builtin_nontemporal_load(p0 + 1);
  v4f c = __builtin_nontemporal_load(p1);
  v4f d = __builtin_nontemporal_load(p1 + 1);
  v16h r;
#pragma unroll
  for (int i = 0; i < 4; ++i) {
    r[i]      = (_Float16)a[i];
    r[4 + i]  = (_Float16)b[i];
    r[8 + i]  = (_Float16)c[i];
    r[12 + i] = (_Float16)d[i];
  }
  return r;
}

// ---- A operand from LDS f16 row (two ds_load_b128) ----
static __device__ __forceinline__ v16h ld_a_lds(const _Float16* row, int koff, int hi) {
  V16H t;
  t.u[0] = *reinterpret_cast<const uint4*>(row + koff + hi * 8);
  t.u[1] = *reinterpret_cast<const uint4*>(row + koff + 16 + hi * 8);
  return t.v;
}

static __device__ __forceinline__ v8f wmma(v16h a, v16h b, v8f c) {
  return __builtin_amdgcn_wmma_f32_16x16x32_f16(false, a, false, b, (short)0, c, false, false);
}

// ================= weight packing =================
__global__ void pack_b_kernel(const float* __restrict__ src, _Float16* __restrict__ dst,
                              int K, int N) {
  int tid = blockIdx.x * blockDim.x + threadIdx.x;
  int ncN = N >> 4;
  int nChunks = (K >> 5) * ncN;
  int chunk = tid >> 5;
  if (chunk >= nChunks) return;
  int lane = tid & 31;
  int kc = chunk / ncN, nc = chunk - kc * ncN;
  int n = nc * 16 + (lane & 15);
  int kbase = kc * 32 + ((lane >> 4) << 3);
  _Float16* d = dst + (size_t)chunk * 512 + lane * 16;
#pragma unroll
  for (int e = 0; e < 16; ++e)
    d[e] = (_Float16)src[(size_t)(kbase + e + ((e >> 3) << 3)) * N + n];
}

__global__ void pack_w1cat_kernel(const float* __restrict__ radw1,
                                  const float* __restrict__ sphw1,
                                  _Float16* __restrict__ dst) {
  // virtual K=512 (rows 0..255 rad_w1, 256..511 sph_w1), N=128
  int tid = blockIdx.x * blockDim.x + threadIdx.x;
  int chunk = tid >> 5;
  if (chunk >= 16 * 8) return;
  int lane = tid & 31;
  int kc = chunk >> 3, nc = chunk & 7;
  int n = nc * 16 + (lane & 15);
  int kbase = kc * 32 + ((lane >> 4) << 3);
  _Float16* d = dst + (size_t)chunk * 512 + lane * 16;
#pragma unroll
  for (int e = 0; e < 16; ++e) {
    int k = kbase + e + ((e >> 3) << 3);
    float v = (k < 256) ? radw1[(size_t)k * DD + n] : sphw1[(size_t)(k - 256) * DD + n];
    d[e] = (_Float16)v;
  }
}

__global__ void zero_kernel(float* __restrict__ p, int n) {
  int i = blockIdx.x * blockDim.x + threadIdx.x;
  if (i < n) p[i] = 0.0f;
}

// ================= q/k/v projections: node_feats @ {wq,wk,wv} =================
static __device__ __forceinline__ void proj16(const v16h* aN, const _Float16* Wp,
                                              float* __restrict__ O,
                                              int base, int hi, int ln, int lane) {
#pragma unroll
  for (int nc = 0; nc < 8; ++nc) {
    // batch the 4 B-chunk loads, then 4 back-to-back WMMAs
    v16h b[4];
#pragma unroll
    for (int kc = 0; kc < 4; ++kc) b[kc] = ld_b(Wp, kc * 8 + nc, lane);
    v8f acc = {};
#pragma unroll
    for (int kc = 0; kc < 4; ++kc) acc = wmma(aN[kc], b[kc], acc);
#pragma unroll
    for (int v = 0; v < 8; ++v)
      O[(size_t)(base + v + 8 * hi) * DD + nc * 16 + ln] = acc[v];
  }
}

__global__ __launch_bounds__(NW * 32) void qkv_kernel(
    const float* __restrict__ node_feats,
    const _Float16* __restrict__ wqp, const _Float16* __restrict__ wkp,
    const _Float16* __restrict__ wvp,
    float* __restrict__ qbuf, float* __restrict__ kbuf, float* __restrict__ vbuf) {
  const int lane = threadIdx.x & 31;
  const int w = threadIdx.x >> 5;
  const int hi = lane >> 4;
  const int ln = lane & 15;
  const int tile = blockIdx.x * NW + w;
  if (tile >= NN / 16) return;   // wave-uniform
  const int base = tile * 16;

  const float* nrow = node_feats + (size_t)(base + ln) * DD;
  v16h aN[4];
#pragma unroll
  for (int kc = 0; kc < 4; ++kc) aN[kc] = ld_a_f32_nt(nrow, kc * 32, hi);

  proj16(aN, wqp, qbuf, base, hi, ln, lane);
  proj16(aN, wkp, kbuf, base, hi, ln, lane);
  proj16(aN, wvp, vbuf, base, hi, ln, lane);
}

// ================= fused edge kernel =================
__global__ __launch_bounds__(NW * 32) void edge_kernel(
    const float* __restrict__ edge_feats, const float* __restrict__ chi,
    const float* __restrict__ cutoffs, const int* __restrict__ senders,
    const int* __restrict__ receivers,
    const float* __restrict__ rad_b0, const float* __restrict__ sph_b0,
    const float* __restrict__ rad_b1, const float* __restrict__ sph_b1,
    const _Float16* __restrict__ w0rad_p, const _Float16* __restrict__ w0sph_p,
    const _Float16* __restrict__ w1cat_p,
    const float* __restrict__ qbuf, const float* __restrict__ kbuf,
    const float* __restrict__ vbuf, float* __restrict__ out) {
  __shared__ __align__(16) _Float16 hid[NW][16][HIDN];  // 32 KB / block

  const int lane = threadIdx.x & 31;
  const int w = threadIdx.x >> 5;
  const int hi = lane >> 4;
  const int ln = lane & 15;
  const int tile = blockIdx.x * NW + w;   // exact: 100000 tiles
  const int base = tile * 16;

  // per-edge metadata (single-use -> non-temporal, broadcast within 16-lane half)
  int recvI[8], sendI[8];
  float cutV[8];
#pragma unroll
  for (int v = 0; v < 8; ++v) {
    int e = base + v + 8 * hi;
    recvI[v] = __builtin_nontemporal_load(&receivers[e]);
    sendI[v] = __builtin_nontemporal_load(&senders[e]);
    cutV[v]  = __builtin_nontemporal_load(&cutoffs[e]);
  }

  // speculative prefetch of next tile's edge features
  {
    size_t pe = (size_t)(base + 16 + ln);
    if (pe >= NE) pe = NE - 1;
    __builtin_prefetch(edge_feats + pe * RADIN, 0, 0);
  }

  // -------- layer 1 (rad): hidden = silu(edge_feats @ rad_w0 + b0) --------
  const float* erow = edge_feats + (size_t)(base + ln) * RADIN;
  v16h aRad0 = ld_a_f32_nt(erow, 0, hi);
  v16h aRad1 = ld_a_f32_nt(erow, 32, hi);
#pragma unroll
  for (int nc = 0; nc < 16; ++nc) {
    v16h b0c = ld_b(w0rad_p, 0 * 16 + nc, lane);
    v16h b1c = ld_b(w0rad_p, 1 * 16 + nc, lane);
    v8f acc = {};
    acc = wmma(aRad0, b0c, acc);
    acc = wmma(aRad1, b1c, acc);
    float b = rad_b0[nc * 16 + ln];
#pragma unroll
    for (int v = 0; v < 8; ++v) {
      float x = acc[v] + b;
      float s = x / (1.0f + __expf(-x));          // silu
      hid[w][v + 8 * hi][nc * 16 + ln] = (_Float16)s;
    }
  }

  // -------- layer 2, rad half: accT += hidden_rad @ rad_w1 --------
  v8f accT[8];
#pragma unroll
  for (int t = 0; t < 8; ++t) { v8f z = {}; accT[t] = z; }
  const _Float16* hrow = &hid[w][ln][0];
#pragma unroll
  for (int kc = 0; kc < 8; ++kc) {
    v16h aH = ld_a_lds(hrow, kc * 32, hi);
    // one load clause (16 x b128) then 8 back-to-back WMMAs
    v16h b[8];
#pragma unroll
    for (int nc = 0; nc < 8; ++nc) b[nc] = ld_b(w1cat_p, kc * 8 + nc, lane);
#pragma unroll
    for (int nc = 0; nc < 8; ++nc) accT[nc] = wmma(aH, b[nc], accT[nc]);
  }

  // -------- layer 1 (sph), reusing the LDS slab --------
  const float* crow = chi + (size_t)(base + ln) * SPHIN;
  v16h aSph = ld_a_f32_nt(crow, 0, hi);
#pragma unroll
  for (int nc = 0; nc < 16; ++nc) {
    v8f acc = {};
    acc = wmma(aSph, ld_b(w0sph_p, nc, lane), acc);
    float b = sph_b0[nc * 16 + ln];
#pragma unroll
    for (int v = 0; v < 8; ++v) {
      float x = acc[v] + b;
      float s = x / (1.0f + __expf(-x));
      hid[w][v + 8 * hi][nc * 16 + ln] = (_Float16)s;
    }
  }

  // -------- layer 2, sph half: accT += hidden_sph @ sph_w1 --------
#pragma unroll
  for (int kc = 0; kc < 8; ++kc) {
    v16h aH = ld_a_lds(hrow, kc * 32, hi);
    v16h b[8];
#pragma unroll
    for (int nc = 0; nc < 8; ++nc) b[nc] = ld_b(w1cat_p, (kc + 8) * 8 + nc, lane);
#pragma unroll
    for (int nc = 0; nc < 8; ++nc) accT[nc] = wmma(aH, b[nc], accT[nc]);
  }

  // -------- attention epilogue: tile nc == head nc; lane ln == channel-in-head --------
#pragma unroll
  for (int nc = 0; nc < 8; ++nc) {
    int ch = nc * 16 + ln;
    float b1 = rad_b1[ch] + sph_b1[ch];
#pragma unroll
    for (int v = 0; v < 8; ++v) {
      float wv_ = accT[nc][v] + b1;
      int nr = recvI[v], ns = sendI[v];
      float qv = qbuf[(size_t)nr * DD + ch];   // L2-resident (RT hints)
      float kv = kbuf[(size_t)ns * DD + ch];
      float vv = vbuf[(size_t)ns * DD + ch];
      float p = qv * wv_ * kv;
      // reduce over the 16 channels of this head (stays within each 16-lane half)
      p += __shfl_xor(p, 1, 32);
      p += __shfl_xor(p, 2, 32);
      p += __shfl_xor(p, 4, 32);
      p += __shfl_xor(p, 8, 32);
      // fold 1/sqrt(16) and 1/AVG_NUM_NEIGHBORS(16) -> 1/64
      float alpha = p * cutV[v] * (1.0f / 64.0f);
      atomicAdd(&out[(size_t)nr * DD + ch], alpha * vv);
    }
  }
}

// ================= host launcher =================
extern "C" void kernel_launch(void* const* d_in, const int* in_sizes, int n_in,
                              void* d_out, int out_size, void* d_ws, size_t ws_size,
                              hipStream_t stream) {
  const float* node_feats = (const float*)d_in[0];
  const float* edge_feats = (const float*)d_in[1];
  const float* chi        = (const float*)d_in[2];
  const float* cutoffs    = (const float*)d_in[3];
  const int*   senders    = (const int*)d_in[4];
  const int*   receivers  = (const int*)d_in[5];
  const float* rad_w0 = (const float*)d_in[6];
  const float* rad_b0 = (const float*)d_in[7];
  const float* rad_w1 = (const float*)d_in[8];
  const float* rad_b1 = (const float*)d_in[9];
  const float* sph_w0 = (const float*)d_in[10];
  const float* sph_b0 = (const float*)d_in[11];
  const float* sph_w1 = (const float*)d_in[12];
  const float* sph_b1 = (const float*)d_in[13];
  const float* wq = (const float*)d_in[14];
  const float* wk = (const float*)d_in[15];
  const float* wv = (const float*)d_in[16];
  float* out = (float*)d_out;

  char* ws = (char*)d_ws;
  const size_t szQ = (size_t)NN * DD * sizeof(float);
  float* qbuf = (float*)ws;
  float* kbuf = (float*)(ws + szQ);
  float* vbuf = (float*)(ws + 2 * szQ);
  _Float16* w0rad_p = (_Float16*)(ws + 3 * szQ);   //  32 chunks
  _Float16* w0sph_p = w0rad_p + 32 * 512;          //  16 chunks
  _Float16* w1cat_p = w0sph_p + 16 * 512;          // 128 chunks
  _Float16* wq_p    = w1cat_p + 128 * 512;         //  32 chunks each
  _Float16* wk_p    = wq_p + 32 * 512;
  _Float16* wv_p    = wk_p + 32 * 512;

  // 1) pack weights into WMMA-B f16 layout
  pack_b_kernel<<<(32 * 32 + 255) / 256, 256, 0, stream>>>(rad_w0, w0rad_p, RADIN, HIDN);
  pack_b_kernel<<<(16 * 32 + 255) / 256, 256, 0, stream>>>(sph_w0, w0sph_p, SPHIN, HIDN);
  pack_w1cat_kernel<<<(128 * 32 + 255) / 256, 256, 0, stream>>>(rad_w1, sph_w1, w1cat_p);
  pack_b_kernel<<<(32 * 32 + 255) / 256, 256, 0, stream>>>(wq, wq_p, DD, DD);
  pack_b_kernel<<<(32 * 32 + 255) / 256, 256, 0, stream>>>(wk, wk_p, DD, DD);
  pack_b_kernel<<<(32 * 32 + 255) / 256, 256, 0, stream>>>(wv, wv_p, DD, DD);

  // 2) zero output (we accumulate with atomics)
  zero_kernel<<<(NN * DD + 255) / 256, 256, 0, stream>>>(out, NN * DD);

  // 3) q/k/v projections (WMMA)
  {
    int tiles = NN / 16;                    // 6250
    int blocks = (tiles + NW - 1) / NW;     // 1563
    qkv_kernel<<<blocks, NW * 32, 0, stream>>>(node_feats, wq_p, wk_p, wv_p,
                                               qbuf, kbuf, vbuf);
  }

  // 4) fused edge MLPs + attention + scatter (WMMA + LDS + atomics)
  {
    int blocks = NE / (16 * NW);            // 25000, exact
    edge_kernel<<<blocks, NW * 32, 0, stream>>>(
        edge_feats, chi, cutoffs, senders, receivers,
        rad_b0, sph_b0, rad_b1, sph_b1,
        w0rad_p, w0sph_p, w1cat_p,
        qbuf, kbuf, vbuf, out);
  }
}
